// SparseMoELayer_49374944034995
// MI455X (gfx1250) — matrix operand order repaired and checked by reading, not compiled, not run on hardware
//
#include <hip/hip_runtime.h>

#define T_TOK    4096
#define DM       768
#define DF       1536
#define NE       8

typedef __bf16 bf16_t;
typedef __attribute__((ext_vector_type(16))) __bf16 v16bf;
typedef __attribute__((ext_vector_type(8)))  __bf16 v8bf;
typedef __attribute__((ext_vector_type(8)))  float  v8f;

__device__ __forceinline__ bf16_t f2bf(float f) {
  unsigned u = __builtin_bit_cast(unsigned, f);
  u += 0x7fffu + ((u >> 16) & 1u);              // round-to-nearest-even
  unsigned short s = (unsigned short)(u >> 16);
  return __builtin_bit_cast(bf16_t, s);
}

// A fragment (16x32 bf16): lane holds row m=lane&15, K in {hi*16..hi*16+15}
__device__ __forceinline__ v16bf load_afrag(const bf16_t* __restrict__ row, int k0, int hi) {
  v8bf a0 = *(const v8bf*)(row + k0 + hi * 16);
  v8bf a1 = *(const v8bf*)(row + k0 + hi * 16 + 8);
  v16bf af;
#pragma unroll
  for (int q = 0; q < 8; q++) { af[q] = a0[q]; af[q + 8] = a1[q]; }
  return af;
}

// ---------------- zero per-expert counters ----------------
__global__ void k_zero(int* cnt) {
  if (threadIdx.x < NE) cnt[threadIdx.x] = 0;
}

// ---------------- fp32 -> bf16 weight conversion ----------------
__global__ void k_cvt(const float* __restrict__ src, bf16_t* __restrict__ dst, int n) {
  int i = blockIdx.x * blockDim.x + threadIdx.x;
  int stride = gridDim.x * blockDim.x;
  for (; i < n; i += stride) dst[i] = f2bf(src[i]);
}

// ---------------- RMSNorm + router + top-2 routing ----------------
__global__ __launch_bounds__(256) void k_router(
    const float* __restrict__ x, const float* __restrict__ norm_w,
    const float* __restrict__ router_w,
    bf16_t* __restrict__ xnbf, float* __restrict__ combw,
    int* __restrict__ assign, int* __restrict__ cnt)
{
  __shared__ float s_xn[DM];
  __shared__ float s_red[256];
  __shared__ float s_logit[NE];
  int t = blockIdx.x;
  int tid = threadIdx.x;
  const float* xr = x + (size_t)t * DM;

  float v0 = xr[tid], v1 = xr[tid + 256], v2 = xr[tid + 512];
  s_red[tid] = v0 * v0 + v1 * v1 + v2 * v2;
  __syncthreads();
  for (int s = 128; s > 0; s >>= 1) {
    if (tid < s) s_red[tid] += s_red[tid + s];
    __syncthreads();
  }
  float rinv = rsqrtf(s_red[0] / (float)DM + 1e-6f);
  {
    float a0 = v0 * rinv * norm_w[tid];
    float a1 = v1 * rinv * norm_w[tid + 256];
    float a2 = v2 * rinv * norm_w[tid + 512];
    s_xn[tid] = a0; s_xn[tid + 256] = a1; s_xn[tid + 512] = a2;
    bf16_t* o = xnbf + (size_t)t * DM;
    o[tid] = f2bf(a0); o[tid + 256] = f2bf(a1); o[tid + 512] = f2bf(a2);
  }
  __syncthreads();
  // router logits: 8 groups of 32 lanes, one expert each (fp32, full precision)
  {
    int e = tid >> 5, l = tid & 31;
    const float* rw = router_w + e * DM;
    float p = 0.f;
    for (int k = l; k < DM; k += 32) p += s_xn[k] * rw[k];
    s_red[tid] = p;
    __syncthreads();
    for (int s = 16; s > 0; s >>= 1) {
      if (l < s) s_red[tid] += s_red[tid + s];
      __syncthreads();
    }
    if (l == 0) s_logit[e] = s_red[tid];
  }
  __syncthreads();
  if (tid == 0) {
    float mx = s_logit[0];
    for (int e = 1; e < NE; e++) mx = fmaxf(mx, s_logit[e]);
    float p[NE], se = 0.f;
    for (int e = 0; e < NE; e++) { p[e] = __expf(s_logit[e] - mx); se += p[e]; }
    float inv = 1.f / se;
    int i1 = 0;
    for (int e = 1; e < NE; e++) if (p[e] > p[i1]) i1 = e;
    int i2 = (i1 == 0) ? 1 : 0;
    for (int e = 0; e < NE; e++) if (e != i1 && p[e] > p[i2]) i2 = e;
    combw[t * 2 + 0] = p[i1] * inv;   // raw softmax prob (not renormalized over top-k)
    combw[t * 2 + 1] = p[i2] * inv;
    int pos = atomicAdd(&cnt[i1], 1); assign[i1 * T_TOK + pos] = t * 2;
    pos     = atomicAdd(&cnt[i2], 1); assign[i2 * T_TOK + pos] = t * 2 + 1;
  }
}

// ---------------- GEMM1: h = silu(Xg @ Wg^T) * (Xg @ Wu^T) * combw ----------------
// block: 64 gathered tokens x 128 F-columns; 8 waves, each wave one 16-col tile,
// 4 m-fragments, B fragment reused across the 4 WMMAs (4x weight reuse vs M=16).
__global__ __launch_bounds__(256) void k_gemm1(
    const bf16_t* __restrict__ xnbf, const bf16_t* __restrict__ wg,
    const bf16_t* __restrict__ wu, const float* __restrict__ combw,
    const int* __restrict__ assign, const int* __restrict__ cnt,
    bf16_t* __restrict__ hbf)
{
  int e = blockIdx.z;
  int nrow = cnt[e];
  int row0 = blockIdx.x * 64;
  if (row0 >= nrow) return;
  __shared__ int   s_a[64];
  __shared__ float s_w[64];
  int tid = threadIdx.x;
  if (tid < 64) {
    int i = row0 + tid;
    int a = assign[e * T_TOK + ((i < nrow) ? i : (nrow - 1))];
    s_a[tid] = a;
    s_w[tid] = combw[a];
  }
  __syncthreads();
  int wave = tid >> 5, lane = tid & 31;
  int hi = lane >> 4, mn = lane & 15;
  int fcol = blockIdx.y * 128 + wave * 16 + mn;     // this lane's B column (weight row)
  const bf16_t* ar0 = xnbf + (size_t)(s_a[mn]      >> 1) * DM;
  const bf16_t* ar1 = xnbf + (size_t)(s_a[16 + mn] >> 1) * DM;
  const bf16_t* ar2 = xnbf + (size_t)(s_a[32 + mn] >> 1) * DM;
  const bf16_t* ar3 = xnbf + (size_t)(s_a[48 + mn] >> 1) * DM;
  const bf16_t* bgrow = wg + ((size_t)e * DF + fcol) * DM + hi * 16;
  const bf16_t* burow = wu + ((size_t)e * DF + fcol) * DM + hi * 16;
  v8f cg0 = {}, cg1 = {}, cg2 = {}, cg3 = {};
  v8f cu0 = {}, cu1 = {}, cu2 = {}, cu3 = {};
  for (int k0 = 0; k0 < DM; k0 += 32) {
    v16bf bg = *(const v16bf*)(bgrow + k0);
    v16bf bu = *(const v16bf*)(burow + k0);
    v16bf a0 = load_afrag(ar0, k0, hi);
    v16bf a1 = load_afrag(ar1, k0, hi);
    v16bf a2 = load_afrag(ar2, k0, hi);
    v16bf a3 = load_afrag(ar3, k0, hi);
    cg0 = __builtin_amdgcn_wmma_f32_16x16x32_bf16(false, a0, false, bg, (short)0, cg0, false, false);
    cg1 = __builtin_amdgcn_wmma_f32_16x16x32_bf16(false, a1, false, bg, (short)0, cg1, false, false);
    cg2 = __builtin_amdgcn_wmma_f32_16x16x32_bf16(false, a2, false, bg, (short)0, cg2, false, false);
    cg3 = __builtin_amdgcn_wmma_f32_16x16x32_bf16(false, a3, false, bg, (short)0, cg3, false, false);
    cu0 = __builtin_amdgcn_wmma_f32_16x16x32_bf16(false, a0, false, bu, (short)0, cu0, false, false);
    cu1 = __builtin_amdgcn_wmma_f32_16x16x32_bf16(false, a1, false, bu, (short)0, cu1, false, false);
    cu2 = __builtin_amdgcn_wmma_f32_16x16x32_bf16(false, a2, false, bu, (short)0, cu2, false, false);
    cu3 = __builtin_amdgcn_wmma_f32_16x16x32_bf16(false, a3, false, bu, (short)0, cu3, false, false);
  }
  size_t fb = (size_t)blockIdx.y * 128 + wave * 16 + mn;
  v8f cgs[4] = {cg0, cg1, cg2, cg3};
  v8f cus[4] = {cu0, cu1, cu2, cu3};
#pragma unroll
  for (int mt = 0; mt < 4; mt++) {
#pragma unroll
    for (int j = 0; j < 8; j++) {
      int m = mt * 16 + j + 8 * hi;
      if (row0 + m < nrow) {
        float g = cgs[mt][j], u = cus[mt][j];
        float hval = (g / (1.f + __expf(-g))) * u * s_w[m];
        hbf[(size_t)s_a[m] * DF + fb] = f2bf(hval);
      }
    }
  }
}

// ---------------- GEMM2: y[t,slot,:] = h @ down_w[e]^T ----------------
// block: 64 assignment rows x 128 D-columns.
__global__ __launch_bounds__(256) void k_gemm2(
    const bf16_t* __restrict__ hbf, const bf16_t* __restrict__ wd,
    const int* __restrict__ assign, const int* __restrict__ cnt,
    float* __restrict__ yf)
{
  int e = blockIdx.z;
  int nrow = cnt[e];
  int row0 = blockIdx.x * 64;
  if (row0 >= nrow) return;
  __shared__ int s_a[64];
  int tid = threadIdx.x;
  if (tid < 64) {
    int i = row0 + tid;
    s_a[tid] = assign[e * T_TOK + ((i < nrow) ? i : (nrow - 1))];
  }
  __syncthreads();
  int wave = tid >> 5, lane = tid & 31;
  int hi = lane >> 4, mn = lane & 15;
  int dcol = blockIdx.y * 128 + wave * 16 + mn;
  const bf16_t* ar0 = hbf + (size_t)s_a[mn]      * DF;
  const bf16_t* ar1 = hbf + (size_t)s_a[16 + mn] * DF;
  const bf16_t* ar2 = hbf + (size_t)s_a[32 + mn] * DF;
  const bf16_t* ar3 = hbf + (size_t)s_a[48 + mn] * DF;
  const bf16_t* brow = wd + ((size_t)e * DM + dcol) * DF + hi * 16;
  v8f c0 = {}, c1 = {}, c2 = {}, c3 = {};
  for (int k0 = 0; k0 < DF; k0 += 32) {
    v16bf bb = *(const v16bf*)(brow + k0);
    v16bf a0 = load_afrag(ar0, k0, hi);
    v16bf a1 = load_afrag(ar1, k0, hi);
    v16bf a2 = load_afrag(ar2, k0, hi);
    v16bf a3 = load_afrag(ar3, k0, hi);
    c0 = __builtin_amdgcn_wmma_f32_16x16x32_bf16(false, a0, false, bb, (short)0, c0, false, false);
    c1 = __builtin_amdgcn_wmma_f32_16x16x32_bf16(false, a1, false, bb, (short)0, c1, false, false);
    c2 = __builtin_amdgcn_wmma_f32_16x16x32_bf16(false, a2, false, bb, (short)0, c2, false, false);
    c3 = __builtin_amdgcn_wmma_f32_16x16x32_bf16(false, a3, false, bb, (short)0, c3, false, false);
  }
  v8f cs[4] = {c0, c1, c2, c3};
#pragma unroll
  for (int mt = 0; mt < 4; mt++) {
#pragma unroll
    for (int j = 0; j < 8; j++) {
      int m = mt * 16 + j + 8 * hi;
      if (row0 + m < nrow)
        yf[(size_t)s_a[m] * DM + dcol] = cs[mt][j];
    }
  }
}

// ---------------- finalize: out = residual + y[t,0] + y[t,1] ----------------
__global__ void k_final(const float* __restrict__ x, const float* __restrict__ yf,
                        float* __restrict__ out, int n) {
  int i = blockIdx.x * blockDim.x + threadIdx.x;
  int stride = gridDim.x * blockDim.x;
  for (; i < n; i += stride) {
    int t = i / DM, d = i - t * DM;
    out[i] = x[i] + yf[(size_t)(t * 2) * DM + d] + yf[(size_t)(t * 2 + 1) * DM + d];
  }
}

extern "C" void kernel_launch(void* const* d_in, const int* in_sizes, int n_in,
                              void* d_out, int out_size, void* d_ws, size_t ws_size,
                              hipStream_t stream) {
  (void)in_sizes; (void)n_in; (void)out_size; (void)ws_size;
  const float* x        = (const float*)d_in[0];
  const float* norm_w   = (const float*)d_in[1];
  const float* router_w = (const float*)d_in[2];
  const float* gate_w   = (const float*)d_in[3];
  const float* up_w     = (const float*)d_in[4];
  const float* down_w   = (const float*)d_in[5];
  float* out = (float*)d_out;

  char* p = (char*)d_ws;
  auto carve = [&](size_t bytes) -> char* {
    char* r = p;
    p += (bytes + 255) & ~(size_t)255;
    return r;
  };
  const size_t WELEMS = (size_t)NE * DF * DM;                 // 9.44M per tensor
  bf16_t* wg    = (bf16_t*)carve(WELEMS * 2);
  bf16_t* wu    = (bf16_t*)carve(WELEMS * 2);
  bf16_t* wd    = (bf16_t*)carve(WELEMS * 2);
  bf16_t* xnbf  = (bf16_t*)carve((size_t)T_TOK * DM * 2);
  bf16_t* hbf   = (bf16_t*)carve((size_t)T_TOK * 2 * DF * 2);
  float*  yf    = (float*)carve((size_t)T_TOK * 2 * DM * 4);
  float*  combw = (float*)carve((size_t)T_TOK * 2 * 4);
  int*    assign= (int*)carve((size_t)NE * T_TOK * 4);
  int*    cnt   = (int*)carve(64);

  k_zero<<<1, 32, 0, stream>>>(cnt);
  k_cvt<<<4096, 256, 0, stream>>>(gate_w, wg, (int)WELEMS);
  k_cvt<<<4096, 256, 0, stream>>>(up_w,   wu, (int)WELEMS);
  k_cvt<<<4096, 256, 0, stream>>>(down_w, wd, (int)WELEMS);
  k_router<<<T_TOK, 256, 0, stream>>>(x, norm_w, router_w, xnbf, combw, assign, cnt);
  k_gemm1<<<dim3(T_TOK / 64, DF / 128, NE), 256, 0, stream>>>(xnbf, wg, wu, combw, assign, cnt, hbf);
  k_gemm2<<<dim3(T_TOK / 64, DM / 128, NE), 256, 0, stream>>>(hbf, wd, assign, cnt, yf);
  k_final<<<2048, 256, 0, stream>>>(x, yf, out, T_TOK * DM);
}
